// enhance_net_nopool_28698971472534
// MI455X (gfx1250) — matrix operand order repaired and checked by reading, not compiled, not run on hardware
//
#include <hip/hip_runtime.h>
#include <hip/hip_bf16.h>

typedef __attribute__((ext_vector_type(16))) _Float16 v16h;
typedef __attribute__((ext_vector_type(8)))  _Float16 h8;
typedef __attribute__((ext_vector_type(8)))  float    v8f;

#define HWS 262144L /* 512*512 */
#define NBATCH 8

// ===================== elementwise: v = (r+g+b)/3 (float4) =====================
__global__ __launch_bounds__(256) void k_compute_v(const float* __restrict__ x,
                                                   float* __restrict__ v) {
  long i4 = (long)blockIdx.x * blockDim.x + threadIdx.x;
  if (i4 >= NBATCH * HWS / 4) return;
  long i = i4 * 4;
  int b = (int)(i / HWS);
  long p = i - (long)b * HWS;
  const float4 r  = *(const float4*)(x + ((long)b * 3 + 0) * HWS + p);
  const float4 g  = *(const float4*)(x + ((long)b * 3 + 1) * HWS + p);
  const float4 bl = *(const float4*)(x + ((long)b * 3 + 2) * HWS + p);
  float4 o;
  o.x = (r.x + g.x + bl.x) * (1.f / 3.f);
  o.y = (r.y + g.y + bl.y) * (1.f / 3.f);
  o.z = (r.z + g.z + bl.z) * (1.f / 3.f);
  o.w = (r.w + g.w + bl.w) * (1.f / 3.f);
  *(float4*)(v + (long)b * HWS + p) = o;
}

// ============ weight convert + zero-pad: (N,K) f32 -> (N,KP) f16 ============
__global__ __launch_bounds__(256) void k_pad_w_f16(const float* __restrict__ w,
                                                   _Float16* __restrict__ wh,
                                                   int N, int K, int KP) {
  int i = blockIdx.x * blockDim.x + threadIdx.x;
  if (i >= N * KP) return;
  int n = i / KP, k = i - n * KP;
  wh[i] = (k < K) ? (_Float16)w[n * K + k] : (_Float16)0.f;
}

// ===================== 3x3 conv, pad 1, fused activation =====================
// ACT 0 = ReLU, 1 = sigmoid. Input channels = CINA (from inA) + CINB (from inB).
template <int CINA, int CINB, int COUT, int ACT>
__global__ __launch_bounds__(256) void k_conv3x3(const float* __restrict__ inA,
                                                 const float* __restrict__ inB,
                                                 const float* __restrict__ wgt,
                                                 const float* __restrict__ bias,
                                                 float* __restrict__ out) {
  const int H = 512, W = 512;
  const int CIN = CINA + CINB;
  __shared__ float sw[COUT * CIN * 9];
  __shared__ float sb[COUT];
  for (int i = threadIdx.x; i < COUT * CIN * 9; i += blockDim.x) sw[i] = wgt[i];
  if (threadIdx.x < COUT) sb[threadIdx.x] = bias[threadIdx.x];
  __syncthreads();

  long idx = (long)blockIdx.x * blockDim.x + threadIdx.x;
  if (idx >= NBATCH * HWS) return;
  int b = (int)(idx / HWS);
  long p = idx - (long)b * HWS;
  int y = (int)(p / W), xx = (int)(p % W);

  float acc[COUT];
#pragma unroll
  for (int co = 0; co < COUT; ++co) acc[co] = sb[co];

#pragma unroll
  for (int c = 0; c < CIN; ++c) {
    const float* src = (c < CINA) ? (inA + ((long)b * CINA + c) * HWS)
                                  : (inB + ((long)b * CINB + (c - CINA)) * HWS);
    float vals[9];
#pragma unroll
    for (int ky = 0; ky < 3; ++ky) {
#pragma unroll
      for (int kx = 0; kx < 3; ++kx) {
        int yy = y + ky - 1, xc = xx + kx - 1;
        vals[ky * 3 + kx] =
            (yy >= 0 && yy < H && xc >= 0 && xc < W) ? src[(long)yy * W + xc] : 0.f;
      }
    }
#pragma unroll
    for (int co = 0; co < COUT; ++co) {
      const float* wc = &sw[(co * CIN + c) * 9];
#pragma unroll
      for (int t = 0; t < 9; ++t) acc[co] = fmaf(vals[t], wc[t], acc[co]);
    }
  }
#pragma unroll
  for (int co = 0; co < COUT; ++co) {
    float r = acc[co];
    if (ACT == 0) r = fmaxf(r, 0.f);
    else          r = 1.f / (1.f + __expf(-r));
    out[((long)b * COUT + co) * HWS + p] = r;
  }
}

// ===== small path: v32 gather + stride-2 conv + leaky + instance norm ========
// writes src (f32, token-major (s*8+b)*16+c) and src_h (f16, (s*8+b)*32+c, k>=16 zero)
__global__ __launch_bounds__(256) void k_small_path(const float* __restrict__ v,
                                                    const float* __restrict__ m_w,
                                                    const float* __restrict__ m_b,
                                                    const float* __restrict__ in_g,
                                                    const float* __restrict__ in_b,
                                                    float* __restrict__ src,
                                                    _Float16* __restrict__ srch) {
  int b = blockIdx.x;
  int t = threadIdx.x;
  __shared__ float v32[32 * 32];
  __shared__ float hb[16 * 256];
  __shared__ float stats[32];

  for (int i = t; i < 1024; i += 256) {
    int r = i >> 5, c = i & 31;
    v32[i] = v[(long)b * HWS + (long)(r * 16) * 512 + (c * 16)];
  }
  __syncthreads();

  int oy = t >> 4, ox = t & 15;
  float vals[9];
#pragma unroll
  for (int ky = 0; ky < 3; ++ky) {
#pragma unroll
    for (int kx = 0; kx < 3; ++kx) {
      int iy = 2 * oy - 1 + ky, ix = 2 * ox - 1 + kx;
      vals[ky * 3 + kx] =
          (iy >= 0 && iy < 32 && ix >= 0 && ix < 32) ? v32[iy * 32 + ix] : 0.f;
    }
  }
#pragma unroll
  for (int co = 0; co < 16; ++co) {
    float a = m_b[co];
#pragma unroll
    for (int k = 0; k < 9; ++k) a = fmaf(vals[k], m_w[co * 9 + k], a);
    a = (a >= 0.f) ? a : 0.2f * a;
    hb[co * 256 + t] = a;
  }
  __syncthreads();

  if (t < 16) {
    float s = 0.f, s2 = 0.f;
    for (int i = 0; i < 256; ++i) {
      float xv = hb[t * 256 + i];
      s += xv; s2 += xv * xv;
    }
    float mu = s * (1.f / 256.f);
    float var = s2 * (1.f / 256.f) - mu * mu;
    stats[t] = mu;
    stats[16 + t] = rsqrtf(var + 1e-5f);
  }
  __syncthreads();

  long tok = (long)t * 8 + b;
#pragma unroll
  for (int c = 0; c < 16; ++c) {
    float yv = (hb[c * 256 + t] - stats[c]) * stats[16 + c] * in_g[c] + in_b[c];
    src[tok * 16 + c] = yv;
    srch[tok * 32 + c] = (_Float16)yv;
    srch[tok * 32 + 16 + c] = (_Float16)0.f;
  }
}

// ============== WMMA fragment loaders from padded f16 buffers ==============
// A (16x32): lane l holds M = l&15; K = 16*(l>>4) + e  ->  one contiguous 32B chunk.
__device__ inline v16h load_a_frag_h(const _Float16* __restrict__ A, int row0,
                                     int ldp, int kbase, int lane) {
  int m = lane & 15, kb = (lane >> 4) * 16;
  return *(const v16h*)(A + (long)(row0 + m) * ldp + kbase + kb);
}
// B (32x16) = W^T with W stored (N,KP) f16: lane l holds N = l&15;
// K = 16*(l>>4) + e  ->  one contiguous 32B chunk.
__device__ inline v16h load_b_frag_h(const _Float16* __restrict__ W, int n0,
                                     int ldp, int kbase, int lane) {
  int n = lane & 15, kb = (lane >> 4) * 16;
  return *(const v16h*)(W + (long)(n0 + n) * ldp + kbase + kb);
}

// ===================== QKV projection: (2048,16)x(16,48) =====================
__global__ __launch_bounds__(32) void k_qkv_gemm(const _Float16* __restrict__ srch,
                                                 const _Float16* __restrict__ wqh,
                                                 const float* __restrict__ attn_b,
                                                 float* __restrict__ qkv) {
  int lane = threadIdx.x;
  int row0 = blockIdx.x * 16;
  v16h a = load_a_frag_h(srch, row0, 32, 0, lane);
  int n = lane & 15, mo = (lane >> 4) * 8;
#pragma unroll
  for (int nt = 0; nt < 3; ++nt) {
    v16h bfr = load_b_frag_h(wqh, nt * 16, 32, 0, lane);
    v8f c = {};
    c = __builtin_amdgcn_wmma_f32_16x16x32_f16(false, a, false, bfr, (short)0, c,
                                               false, false);
    float bias = attn_b[nt * 16 + n];
#pragma unroll
    for (int r = 0; r < 8; ++r)
      qkv[(long)(row0 + mo + r) * 48 + nt * 16 + n] = c[r] + bias;
  }
}

// ============ attention: per (b,h), S=256, hd=2, softmax in-block ============
// writes padded f16 o_h (2048,32); h==0 blocks also zero the pad columns
__global__ __launch_bounds__(256) void k_attention(const float* __restrict__ qkv,
                                                   _Float16* __restrict__ oh) {
  int b = blockIdx.x >> 3, h = blockIdx.x & 7;
  int s = threadIdx.x;
  __shared__ float kb[256][2];
  __shared__ float vb[256][2];
  long base = ((long)s * 8 + b) * 48;
  float q0 = qkv[base + h * 2], q1 = qkv[base + h * 2 + 1];
  kb[s][0] = qkv[base + 16 + h * 2];
  kb[s][1] = qkv[base + 16 + h * 2 + 1];
  vb[s][0] = qkv[base + 32 + h * 2];
  vb[s][1] = qkv[base + 32 + h * 2 + 1];
  __syncthreads();
  const float scale = 0.70710678118f; // 1/sqrt(2)
  float m = -1e30f;
  for (int t = 0; t < 256; ++t) {
    float d = (q0 * kb[t][0] + q1 * kb[t][1]) * scale;
    m = fmaxf(m, d);
  }
  float den = 0.f, o0 = 0.f, o1 = 0.f;
  for (int t = 0; t < 256; ++t) {
    float d = (q0 * kb[t][0] + q1 * kb[t][1]) * scale;
    float e = __expf(d - m);
    den += e;
    o0 = fmaf(e, vb[t][0], o0);
    o1 = fmaf(e, vb[t][1], o1);
  }
  float inv = 1.f / den;
  long tok = (long)s * 8 + b;
  oh[tok * 32 + h * 2]     = (_Float16)(o0 * inv);
  oh[tok * 32 + h * 2 + 1] = (_Float16)(o1 * inv);
  if (h == 0) {
#pragma unroll
    for (int c = 16; c < 32; ++c) oh[tok * 32 + c] = (_Float16)0.f;
  }
}

// ===================== out projection: (2048,16)x(16,16) =====================
__global__ __launch_bounds__(32) void k_proj_gemm(const _Float16* __restrict__ oh,
                                                  const _Float16* __restrict__ woh,
                                                  const float* __restrict__ out_b,
                                                  float* __restrict__ proj) {
  int lane = threadIdx.x;
  int row0 = blockIdx.x * 16;
  v16h a = load_a_frag_h(oh, row0, 32, 0, lane);
  v16h bfr = load_b_frag_h(woh, 0, 32, 0, lane);
  v8f c = {};
  c = __builtin_amdgcn_wmma_f32_16x16x32_f16(false, a, false, bfr, (short)0, c,
                                             false, false);
  int n = lane & 15, mo = (lane >> 4) * 8;
  float bias = out_b[n];
#pragma unroll
  for (int r = 0; r < 8; ++r)
    proj[(long)(row0 + mo + r) * 16 + n] = c[r] + bias;
}

// ====== residual add + layernorm (E=16); optional padded-f16 mirror ======
__global__ __launch_bounds__(256) void k_add_ln(const float* __restrict__ a,
                                                const float* __restrict__ r,
                                                const float* __restrict__ g,
                                                const float* __restrict__ bb,
                                                float* __restrict__ out,
                                                _Float16* __restrict__ outh) {
  long tok = (long)blockIdx.x * blockDim.x + threadIdx.x;
  if (tok >= 2048) return;
  float xv[16];
  float mu = 0.f;
#pragma unroll
  for (int c = 0; c < 16; ++c) {
    xv[c] = a[tok * 16 + c] + r[tok * 16 + c];
    mu += xv[c];
  }
  mu *= (1.f / 16.f);
  float var = 0.f;
#pragma unroll
  for (int c = 0; c < 16; ++c) {
    float d = xv[c] - mu;
    var = fmaf(d, d, var);
  }
  var *= (1.f / 16.f);
  float rs = rsqrtf(var + 1e-5f);
#pragma unroll
  for (int c = 0; c < 16; ++c) {
    float yv = (xv[c] - mu) * rs * g[c] + bb[c];
    out[tok * 16 + c] = yv;
    if (outh) {
      outh[tok * 32 + c] = (_Float16)yv;
      outh[tok * 32 + 16 + c] = (_Float16)0.f;
    }
  }
}

// ============ FFN1: (2048,16)x(16,128) + ReLU, writes f16 directly ============
__global__ __launch_bounds__(32) void k_ffn1_gemm(const _Float16* __restrict__ srch,
                                                  const _Float16* __restrict__ w1h,
                                                  const float* __restrict__ l1_b,
                                                  _Float16* __restrict__ h) {
  int lane = threadIdx.x;
  int row0 = blockIdx.x * 16;
  v16h a = load_a_frag_h(srch, row0, 32, 0, lane);
  int n = lane & 15, mo = (lane >> 4) * 8;
#pragma unroll
  for (int nt = 0; nt < 8; ++nt) {
    v16h bfr = load_b_frag_h(w1h, nt * 16, 32, 0, lane);
    v8f c = {};
    c = __builtin_amdgcn_wmma_f32_16x16x32_f16(false, a, false, bfr, (short)0, c,
                                               false, false);
    float bias = l1_b[nt * 16 + n];
#pragma unroll
    for (int r = 0; r < 8; ++r)
      h[(long)(row0 + mo + r) * 128 + nt * 16 + n] =
          (_Float16)fmaxf(c[r] + bias, 0.f);
  }
}

// ===================== FFN2: (2048,128)x(128,16), K=4x32 =====================
__global__ __launch_bounds__(32) void k_ffn2_gemm(const _Float16* __restrict__ h,
                                                  const _Float16* __restrict__ w2h,
                                                  const float* __restrict__ l2_b,
                                                  float* __restrict__ f2) {
  int lane = threadIdx.x;
  int row0 = blockIdx.x * 16;
  v8f c = {};
#pragma unroll
  for (int kt = 0; kt < 4; ++kt) {
    v16h a = load_a_frag_h(h, row0, 128, kt * 32, lane);
    v16h bfr = load_b_frag_h(w2h, 0, 128, kt * 32, lane);
    c = __builtin_amdgcn_wmma_f32_16x16x32_f16(false, a, false, bfr, (short)0, c,
                                               false, false);
  }
  int n = lane & 15, mo = (lane >> 4) * 8;
  float bias = l2_b[n];
#pragma unroll
  for (int r = 0; r < 8; ++r)
    f2[(long)(row0 + mo + r) * 16 + n] = c[r] + bias;
}

// ============ level head: full-field 16x16 conv (reduction) + sigmoid ========
__global__ __launch_bounds__(256) void k_level(const float* __restrict__ src2,
                                               const float* __restrict__ f_w,
                                               const float* __restrict__ f_b,
                                               float* __restrict__ level) {
  int b = blockIdx.x >> 1, o = blockIdx.x & 1;
  int t = threadIdx.x; // spatial s = i*16+j
  float acc = 0.f;
#pragma unroll
  for (int c = 0; c < 16; ++c)
    acc = fmaf(src2[((long)t * 8 + b) * 16 + c],
               f_w[((long)(o * 16 + c) * 16 + (t >> 4)) * 16 + (t & 15)], acc);
  __shared__ float red[256];
  red[t] = acc;
  __syncthreads();
  for (int off = 128; off > 0; off >>= 1) {
    if (t < off) red[t] += red[t + off];
    __syncthreads();
  }
  if (t == 0) level[b * 2 + o] = 1.f / (1.f + __expf(-(red[0] + f_b[o])));
}

// ===================== final per-pixel enhance / t (float4) =====================
__global__ __launch_bounds__(256) void k_final(const float* __restrict__ x,
                                               const float* __restrict__ v,
                                               const float* __restrict__ vr,
                                               const float* __restrict__ level,
                                               float* __restrict__ out) {
  long i4 = (long)blockIdx.x * blockDim.x + threadIdx.x;
  if (i4 >= NBATCH * HWS / 4) return;
  long i = i4 * 4;
  int b = (int)(i / HWS);
  long p = i - (long)b * HWS;
  float gg = 0.1f * level[2 * b] + 0.2f;
  float bc = 0.04f * level[2 * b + 1] + 0.06f;
  float lg = log2f(gg);

  const float4 r4  = *(const float4*)(x + ((long)b * 3 + 0) * HWS + p);
  const float4 g4  = *(const float4*)(x + ((long)b * 3 + 1) * HWS + p);
  const float4 b4  = *(const float4*)(x + ((long)b * 3 + 2) * HWS + p);
  const float4 v4  = *(const float4*)(v + (long)b * HWS + p);
  const float4 vr4 = *(const float4*)(vr + (long)b * HWS + p);

  float4 er, eg, eb, tr, tg, tb;
  const float* vv  = (const float*)&v4;
  const float* vrv = (const float*)&vr4;
  const float* rr  = (const float*)&r4;
  const float* gr  = (const float*)&g4;
  const float* br  = (const float*)&b4;
#pragma unroll
  for (int j = 0; j < 4; ++j) {
    float vj = vv[j];
    float v0 = fminf(fmaxf(vj, 1e-6f), 0.999999f);
    float r0 = exp2f(vrv[j] * lg);          // g ^ v_r
    float ev0 = exp2f(r0 * log2f(v0));      // v0 ^ r0
    float d = bc - vj;
    float L = 400.f * d * d * d;
    L = (L < 1e-5f) ? 1e-6f : L;
    float ev = ev0 - L;
    float inv = ev / (vj + 1e-6f);
    float e0 = rr[j] * inv, e1 = gr[j] * inv, e2 = br[j] * inv;
    bool z = vj > 0.04f;
    ((float*)&er)[j] = e0; ((float*)&eg)[j] = e1; ((float*)&eb)[j] = e2;
    ((float*)&tr)[j] = z ? 0.f : e0;
    ((float*)&tg)[j] = z ? 0.f : e1;
    ((float*)&tb)[j] = z ? 0.f : e2;
  }
  *(float4*)(out + ((long)b * 3 + 0) * HWS + p) = er;
  *(float4*)(out + ((long)b * 3 + 1) * HWS + p) = eg;
  *(float4*)(out + ((long)b * 3 + 2) * HWS + p) = eb;
  float* tbase = out + 8388608L; // after enhance(6291456) + v_r(2097152)
  *(float4*)(tbase + ((long)b * 3 + 0) * HWS + p) = tr;
  *(float4*)(tbase + ((long)b * 3 + 1) * HWS + p) = tg;
  *(float4*)(tbase + ((long)b * 3 + 2) * HWS + p) = tb;
}

// ===================== host launcher =====================
extern "C" void kernel_launch(void* const* d_in, const int* in_sizes, int n_in,
                              void* d_out, int out_size, void* d_ws, size_t ws_size,
                              hipStream_t stream) {
  const float* x      = (const float*)d_in[0];
  const float* c1_w   = (const float*)d_in[1];  const float* c1_b = (const float*)d_in[2];
  const float* c2_w   = (const float*)d_in[3];  const float* c2_b = (const float*)d_in[4];
  const float* c3_w   = (const float*)d_in[5];  const float* c3_b = (const float*)d_in[6];
  const float* c4_w   = (const float*)d_in[7];  const float* c4_b = (const float*)d_in[8];
  const float* c5_w   = (const float*)d_in[9];  const float* c5_b = (const float*)d_in[10];
  const float* c6_w   = (const float*)d_in[11]; const float* c6_b = (const float*)d_in[12];
  const float* c7_w   = (const float*)d_in[13]; const float* c7_b = (const float*)d_in[14];
  const float* m_w    = (const float*)d_in[15]; const float* m_b  = (const float*)d_in[16];
  const float* in_g   = (const float*)d_in[17]; const float* in_b = (const float*)d_in[18];
  const float* attn_w = (const float*)d_in[19]; const float* attn_b = (const float*)d_in[20];
  const float* out_w  = (const float*)d_in[21]; const float* out_b  = (const float*)d_in[22];
  const float* l1_w   = (const float*)d_in[23]; const float* l1_b   = (const float*)d_in[24];
  const float* l2_w   = (const float*)d_in[25]; const float* l2_b   = (const float*)d_in[26];
  const float* n1_g   = (const float*)d_in[27]; const float* n1_b   = (const float*)d_in[28];
  const float* n2_g   = (const float*)d_in[29]; const float* n2_b   = (const float*)d_in[30];
  const float* f_w    = (const float*)d_in[31]; const float* f_b    = (const float*)d_in[32];

  float* out = (float*)d_out;
  float* v_r = out + 6291456L; // v_r slice of d_out

  // ---------------- workspace layout (float units) ----------------
  float* ws   = (float*)d_ws;
  float* v    = ws;                    // 8*HWS
  float* x1   = v  + 8 * HWS;          // 32*HWS each below
  float* x2   = x1 + 32 * HWS;
  float* x3   = x2 + 32 * HWS;
  float* x4   = x3 + 32 * HWS;
  float* x5   = x4 + 32 * HWS;
  float* x6   = x5 + 32 * HWS;
  float* srcb = x6 + 32 * HWS;         // 32768  (src tokens f32)
  float* projb= srcb + 32768;          // 32768
  float* src1 = projb + 32768;         // 32768
  float* f2b  = src1 + 32768;          // 32768
  float* src2 = f2b + 32768;           // 32768
  float* qkvb = src2 + 32768;          // 98304
  float* lvl  = qkvb + 98304;          // 16 (padded to 1024)
  float* hbase = lvl + 1024;
  // f16 staging region (counts in f16 elements; 2 per float slot)
  _Float16* srch  = (_Float16*)hbase;              // 2048*32  = 65536 h
  _Float16* obufh = srch  + 65536;                 // 2048*32  = 65536 h
  _Float16* src1h = obufh + 65536;                 // 2048*32  = 65536 h
  _Float16* hbufh = src1h + 65536;                 // 2048*128 = 262144 h
  _Float16* wqh   = hbufh + 262144;                // 48*32  = 1536 h
  _Float16* woh   = wqh + 1536;                    // 16*32  = 512 h
  _Float16* w1h   = woh + 512;                     // 128*32 = 4096 h
  _Float16* w2h   = w1h + 4096;                    // 16*128 = 2048 h

  const long npix = NBATCH * HWS;
  dim3 blk(256);
  dim3 gr_v((unsigned)((npix / 4 + 255) / 256));
  dim3 gr_c((unsigned)((npix + 255) / 256));

  // big path
  k_compute_v<<<gr_v, blk, 0, stream>>>(x, v);
  k_conv3x3<1, 0, 4, 0><<<gr_c, blk, 0, stream>>>(v,  nullptr, c1_w, c1_b, x1);
  k_conv3x3<4, 0, 4, 0><<<gr_c, blk, 0, stream>>>(x1, nullptr, c2_w, c2_b, x2);
  k_conv3x3<4, 0, 4, 0><<<gr_c, blk, 0, stream>>>(x2, nullptr, c3_w, c3_b, x3);
  k_conv3x3<4, 0, 4, 0><<<gr_c, blk, 0, stream>>>(x3, nullptr, c4_w, c4_b, x4);
  k_conv3x3<4, 4, 4, 0><<<gr_c, blk, 0, stream>>>(x3, x4, c5_w, c5_b, x5);
  k_conv3x3<4, 4, 4, 0><<<gr_c, blk, 0, stream>>>(x2, x5, c6_w, c6_b, x6);
  k_conv3x3<4, 4, 1, 1><<<gr_c, blk, 0, stream>>>(x1, x6, c7_w, c7_b, v_r);

  // weight staging to padded f16 (tiny, overlaps with conv work on other WGPs)
  k_pad_w_f16<<<dim3(6),  blk, 0, stream>>>(attn_w, wqh, 48, 16, 32);
  k_pad_w_f16<<<dim3(2),  blk, 0, stream>>>(out_w,  woh, 16, 16, 32);
  k_pad_w_f16<<<dim3(16), blk, 0, stream>>>(l1_w,   w1h, 128, 16, 32);
  k_pad_w_f16<<<dim3(8),  blk, 0, stream>>>(l2_w,   w2h, 16, 128, 128);

  // small path + transformer (WMMA)
  k_small_path<<<dim3(8), blk, 0, stream>>>(v, m_w, m_b, in_g, in_b, srcb, srch);
  k_qkv_gemm<<<dim3(128), dim3(32), 0, stream>>>(srch, wqh, attn_b, qkvb);
  k_attention<<<dim3(64), blk, 0, stream>>>(qkvb, obufh);
  k_proj_gemm<<<dim3(128), dim3(32), 0, stream>>>(obufh, woh, out_b, projb);
  k_add_ln<<<dim3(8), blk, 0, stream>>>(srcb, projb, n1_g, n1_b, src1, src1h);
  k_ffn1_gemm<<<dim3(128), dim3(32), 0, stream>>>(src1h, w1h, l1_b, hbufh);
  k_ffn2_gemm<<<dim3(128), dim3(32), 0, stream>>>(hbufh, w2h, l2_b, f2b);
  k_add_ln<<<dim3(8), blk, 0, stream>>>(src1, f2b, n2_g, n2_b, src2, (_Float16*)nullptr);
  k_level<<<dim3(16), blk, 0, stream>>>(src2, f_w, f_b, lvl);

  // fused per-pixel finale
  k_final<<<gr_v, blk, 0, stream>>>(x, v, v_r, lvl, out);
}
